// GNNLayer_23965917511725
// MI455X (gfx1250) — matrix operand order repaired and checked by reading, not compiled, not run on hardware
//
#include <hip/hip_runtime.h>
#include <hip/hip_bf16.h>

#define N_NODES 100000
#define N_EDGES 1600000
#define IN_F    256
#define OUT_F   128
#define ROWS_PER_BLOCK 64
#define AS_K (IN_F + 8)   // 264 bf16 -> 132 dwords -> bank stride 4 (conflict-free)
#define WT_K (IN_F + 8)

typedef __attribute__((ext_vector_type(16))) __bf16 v16bf;
typedef __attribute__((ext_vector_type(8)))  float  v8f;

static __device__ __forceinline__ unsigned pack2bf(float a, float b) {
    __bf16 x = (__bf16)a, y = (__bf16)b;
    unsigned short ux = __builtin_bit_cast(unsigned short, x);
    unsigned short uy = __builtin_bit_cast(unsigned short, y);
    return (unsigned)ux | ((unsigned)uy << 16);
}

// ---------------------------------------------------------------------------
// Kernel 1: x = (input * drop_mask) @ W   via v_wmma_f32_16x16x32_bf16
// block = 256 threads = 8 waves; block covers 64 rows x 128 cols.
// ---------------------------------------------------------------------------
__global__ __launch_bounds__(256)
void gemm_bf16_wmma(const float* __restrict__ x,
                    const float* __restrict__ mask,
                    const float* __restrict__ w,
                    float* __restrict__ xo)
{
    __shared__ __bf16 As[ROWS_PER_BLOCK][AS_K];  // 33,792 B
    __shared__ __bf16 Wt[OUT_F][WT_K];           // 67,584 B (W transposed, bf16)

    const int tid     = threadIdx.x;
    const int rowBase = blockIdx.x * ROWS_PER_BLOCK;

    // ---- stage W (256x128 f32, row-major) transposed to Wt[col][k] bf16 ----
    // 8192 float4 chunks along N; 32 chunks per thread.
    #pragma unroll 4
    for (int j = 0; j < 32; ++j) {
        int chunk = tid + 256 * j;          // 0..8191
        int k     = chunk >> 5;             // 32 float4 chunks per K-row
        int c     = (chunk & 31) * 4;
        const float4 wv = ((const float4*)(w + k * OUT_F))[chunk & 31];
        Wt[c + 0][k] = (__bf16)wv.x;
        Wt[c + 1][k] = (__bf16)wv.y;
        Wt[c + 2][k] = (__bf16)wv.z;
        Wt[c + 3][k] = (__bf16)wv.w;
    }

    // ---- stage A tile with dropout applied, converted to bf16 ----
    // 64 rows x 256 cols; 4 threads per row, 64 cols (16 float4) each.
    {
        int r  = tid >> 2;
        int cb = (tid & 3) * 64;
        int gr = rowBase + r;
        if (gr >= N_NODES) gr = N_NODES - 1;    // clamp: harmless dup reads
        const float4* xr = (const float4*)(x    + (size_t)gr * IN_F + cb);
        const float4* mr = (const float4*)(mask + (size_t)gr * IN_F + cb);
        #pragma unroll 4
        for (int j = 0; j < 16; ++j) {
            float4 xv = xr[j], mv = mr[j];
            uint2 pk;
            pk.x = pack2bf(xv.x * mv.x, xv.y * mv.y);
            pk.y = pack2bf(xv.z * mv.z, xv.w * mv.w);
            *(uint2*)&As[r][cb + 4 * j] = pk;   // offset multiple of 4 elems -> 8B aligned
        }
    }
    __syncthreads();

    const int wave   = tid >> 5;        // 0..7 -> 16-column tile
    const int lane   = tid & 31;
    const int laneHi = lane >> 4;       // 0/1 half-wave
    const int ln     = lane & 15;
    const int col    = wave * 16 + ln;

    // ---- preload B fragments for all 8 K-steps (reused over 4 row tiles) ----
    // 16-bit B 32x16 layout: lanes 0-15 hold K 0-7 & 16-23, lanes 16-31 hold K 8-15 & 24-31.
    union BV { v16bf v; uint4 q[2]; } bfr[8];
    #pragma unroll
    for (int ks = 0; ks < 8; ++ks) {
        const uint4* p = (const uint4*)&Wt[col][ks * 32 + laneHi * 8];
        bfr[ks].q[0] = p[0];
        bfr[ks].q[1] = p[2];
    }

    // uniform (scalar-branch) guard: only the last block has OOB rows
    const bool fullTile = (rowBase + ROWS_PER_BLOCK) <= N_NODES;

    #pragma unroll
    for (int rt = 0; rt < 4; ++rt) {
        v8f c = {0.f, 0.f, 0.f, 0.f, 0.f, 0.f, 0.f, 0.f};
        const __bf16* arow = &As[rt * 16 + ln][0];
        #pragma unroll
        for (int ks = 0; ks < 8; ++ks) {
            // 16-bit A 16x32 layout: lanes0-15: K 0-7 & 16-23; lanes16-31: K 8-15 & 24-31
            union AV { v16bf v; uint4 q[2]; } a;
            const __bf16* ap = arow + ks * 32 + laneHi * 8;
            a.q[0] = *(const uint4*)(ap);
            a.q[1] = *(const uint4*)(ap + 16);
            c = __builtin_amdgcn_wmma_f32_16x16x32_bf16(
                    false, a.v, false, bfr[ks].v, (short)0, c, false, false);
        }
        // C/D layout: VGPR v -> M = v + laneHi*8, N = ln
        const int r0 = rowBase + rt * 16 + laneHi * 8;
        float* orow = xo + (size_t)r0 * OUT_F + col;
        if (fullTile) {
            // fast path: 8 stores at constant immediate offsets, no exec churn
            #pragma unroll
            for (int v = 0; v < 8; ++v) orow[v * OUT_F] = c[v];
        } else {
            #pragma unroll
            for (int v = 0; v < 8; ++v)
                if (r0 + v < N_NODES) orow[v * OUT_F] = c[v];
        }
    }
}

// ---------------------------------------------------------------------------
// Kernel 2: zero the output accumulator
// ---------------------------------------------------------------------------
__global__ __launch_bounds__(256)
void zero_out(float4* __restrict__ out, int n4)
{
    int i = blockIdx.x * 256 + threadIdx.x;
    if (i < n4) out[i] = make_float4(0.f, 0.f, 0.f, 0.f);
}

// ---------------------------------------------------------------------------
// Kernel 3: edge scatter  out[row[e]] += val[e] * x[col[e]]
// one wave per edge; lane handles a float4 of the 128 features.
// x (51.2MB) and out (51.2MB) are L2-resident (192MB L2) -> L2-bound atomics.
// ---------------------------------------------------------------------------
__global__ __launch_bounds__(256)
void spmm_scatter(const float* __restrict__ xo,
                  const float* __restrict__ val,
                  const int*   __restrict__ erow,
                  const int*   __restrict__ ecol,
                  float* __restrict__ out)
{
    int e    = (blockIdx.x * 256 + threadIdx.x) >> 5;
    int lane = threadIdx.x & 31;
    if (e >= N_EDGES) return;

    int   r = erow[e];
    int   c = ecol[e];
    float v = val[e];

    const float4 xv = *(const float4*)(xo + (size_t)c * OUT_F + lane * 4);
    float* dst = out + (size_t)r * OUT_F + lane * 4;
    unsafeAtomicAdd(dst + 0, v * xv.x);   // global_atomic_add_f32
    unsafeAtomicAdd(dst + 1, v * xv.y);
    unsafeAtomicAdd(dst + 2, v * xv.z);
    unsafeAtomicAdd(dst + 3, v * xv.w);
}

// ---------------------------------------------------------------------------
// Kernel 4: ReLU in place
// ---------------------------------------------------------------------------
__global__ __launch_bounds__(256)
void relu_k(float4* __restrict__ out, int n4)
{
    int i = blockIdx.x * 256 + threadIdx.x;
    if (i < n4) {
        float4 t = out[i];
        t.x = fmaxf(t.x, 0.f); t.y = fmaxf(t.y, 0.f);
        t.z = fmaxf(t.z, 0.f); t.w = fmaxf(t.w, 0.f);
        out[i] = t;
    }
}

extern "C" void kernel_launch(void* const* d_in, const int* in_sizes, int n_in,
                              void* d_out, int out_size, void* d_ws, size_t ws_size,
                              hipStream_t stream)
{
    const float* x    = (const float*)d_in[0];  // input      [100000,256]
    const float* w    = (const float*)d_in[1];  // weight     [256,128]
    const float* adj  = (const float*)d_in[2];  // adj_val    [1.6M]
    const float* mask = (const float*)d_in[3];  // drop_mask  [100000,256]
    const int*   erow = (const int*)d_in[4];    // edge_row
    const int*   ecol = (const int*)d_in[5];    // edge_col
    float* out = (float*)d_out;                 // [100000,128]
    float* xo  = (float*)d_ws;                  // intermediate x: 51.2 MB

    const int n4 = (N_NODES * OUT_F) / 4;       // 3,200,000 float4s

    zero_out<<<(n4 + 255) / 256, 256, 0, stream>>>((float4*)out, n4);
    gemm_bf16_wmma<<<(N_NODES + ROWS_PER_BLOCK - 1) / ROWS_PER_BLOCK, 256, 0, stream>>>(
        x, mask, w, xo);
    spmm_scatter<<<(N_EDGES * 32) / 256, 256, 0, stream>>>(xo, adj, erow, ecol, out);
    relu_k<<<(n4 + 255) / 256, 256, 0, stream>>>((float4*)out, n4);
}
